// ManyChannelsIntegratorNonLinearDecoder_53051436040341
// MI455X (gfx1250) — matrix operand order repaired and checked by reading, not compiled, not run on hardware
//
#include <hip/hip_runtime.h>

typedef __attribute__((ext_vector_type(16))) __bf16 v16bf;
typedef __attribute__((ext_vector_type(8)))  __bf16 v8bf;
typedef __attribute__((ext_vector_type(4)))  __bf16 v4bf;
typedef __attribute__((ext_vector_type(8)))  float  v8f;
typedef __attribute__((ext_vector_type(4)))  float  v4f;

namespace {
constexpr int N_ = 4096, B_ = 64, T_ = 512, C_ = 4;
constexpr int NBLK = 64;          // persistent blocks; each owns TN columns of n
constexpr int TN   = N_ / NBLK;   // 64
constexpr int NTHR = 256;         // 8 x wave32
constexpr float SAT_LO = 0.0f, SAT_HI = 1.0e6f;

// workspace layout (bytes)
constexpr size_t OFF_WBF  = 0;                                    // N*N bf16 (32 MB)
constexpr size_t SZ_WBF   = (size_t)N_ * N_ * 2;
constexpr size_t OFF_ABUF = OFF_WBF + SZ_WBF;                     // B*N bf16
constexpr size_t SZ_ABUF  = (size_t)B_ * N_ * 2;
constexpr size_t OFF_SBUF = OFF_ABUF + SZ_ABUF;                   // B*N f32
constexpr size_t SZ_SBUF  = (size_t)B_ * N_ * 4;
constexpr size_t OFF_DACC = OFF_SBUF + SZ_SBUF;                   // B*C f32
constexpr size_t SZ_DACC  = (size_t)B_ * C_ * 4;
constexpr size_t OFF_BAR  = OFF_DACC + SZ_DACC;                   // 2 x u32
}

__device__ __forceinline__ float satclamp(float x) {
    return fminf(fmaxf(x, SAT_LO), SAT_HI);
}

// Device-wide sense-reversing barrier. All NBLK blocks must be co-resident
// (64 blocks x 256 threads on MI455X: trivially true).
__device__ __forceinline__ void gbar(unsigned* cnt, unsigned* gen) {
    __threadfence();
    __syncthreads();
    if (threadIdx.x == 0) {
        unsigned g = __hip_atomic_load(gen, __ATOMIC_ACQUIRE, __HIP_MEMORY_SCOPE_AGENT);
        unsigned a = __hip_atomic_fetch_add(cnt, 1u, __ATOMIC_ACQ_REL, __HIP_MEMORY_SCOPE_AGENT);
        if (a == (unsigned)NBLK - 1u) {
            __hip_atomic_store(cnt, 0u, __ATOMIC_RELAXED, __HIP_MEMORY_SCOPE_AGENT);
            __hip_atomic_fetch_add(gen, 1u, __ATOMIC_ACQ_REL, __HIP_MEMORY_SCOPE_AGENT);
        } else {
            while (__hip_atomic_load(gen, __ATOMIC_ACQUIRE, __HIP_MEMORY_SCOPE_AGENT) == g)
                __builtin_amdgcn_s_sleep(1);
        }
    }
    __syncthreads();
}

__global__ void rnn_init_kernel(float* dacc, unsigned* barCnt, unsigned* barGen) {
    int i = threadIdx.x;
    if (i < B_ * C_) dacc[i] = 0.0f;
    if (i == 0) { barCnt[0] = 0u; barGen[0] = 0u; }
}

__global__ __launch_bounds__(NTHR)
void rnn_persistent_kernel(const float* __restrict__ inputs,  // [C,B,T]
                           const float* __restrict__ state,   // [B,N]
                           const float* __restrict__ mask,    // [N]
                           const float* __restrict__ W,       // [N,N] row-major, contract over 2nd dim
                           const float* __restrict__ enc,     // [C,N]
                           const float* __restrict__ biasc,   // [N]
                           const float* __restrict__ dec_w,   // [C,N]
                           const float* __restrict__ dec_b,   // [C]
                           float* __restrict__ out,           // [C,B,T]
                           __bf16* __restrict__ Wbf,          // [N,N] bf16
                           __bf16* __restrict__ aBuf,         // [B,N] bf16: u = s + mask*ext
                           float*  __restrict__ sBuf,         // [B,N] f32 state
                           float*  __restrict__ dacc,         // [B,C] decode accumulator
                           unsigned* __restrict__ barCnt,
                           unsigned* __restrict__ barGen)
{
    const int tid = threadIdx.x;
    const int bid = blockIdx.x;
    const int nbase = bid * TN;

    __shared__ float xls[C_ * B_];     // x_t staged per step
    __shared__ float sAcc[B_ * C_];    // per-block decode partials

    // ---- setup: convert our 64 rows of W to bf16 (stays hot in 192MB L2) ----
    {
        const v4f* src = (const v4f*)(W + (size_t)bid * TN * N_);
        v4bf*      dst = (v4bf*)(Wbf + (size_t)bid * TN * N_);
        const int  n4  = TN * N_ / 4;   // 65536
        for (int i = tid; i < n4; i += NTHR) {
            v4f v = src[i];
            v4bf o;
            o[0] = (__bf16)v[0]; o[1] = (__bf16)v[1];
            o[2] = (__bf16)v[2]; o[3] = (__bf16)v[3];
            dst[i] = o;
        }
    }
    // ---- setup: copy initial state (flat 1/NBLK slice per block) ----
    {
        const int per = B_ * N_ / NBLK;  // 4096
        for (int i = tid; i < per; i += NTHR) {
            size_t idx = (size_t)bid * per + i;
            sBuf[idx] = state[idx];
        }
    }
    gbar(barCnt, barGen);

    // wave/lane geometry for the GEMM phase
    const int lane = tid & 31;
    const int wave = tid >> 5;          // 0..7
    const int lr   = lane & 15;
    const int hi   = lane >> 4;         // 0/1
    const int m0   = (wave & 3) * 16;   // batch-row tile base (B=64 -> 4 tiles)
    const int ng   = wave >> 2;         // 0/1 -> pair of n-tiles
    const int n0   = nbase + ng * 32 + lr;
    const int n1   = n0 + 16;

    for (int t = 0; t < T_; ++t) {
        // ================= Phase A: u = s + mask * (x_t . enc) ==============
        if (tid < C_ * B_) {
            int c = tid >> 6, b = tid & 63;
            xls[tid] = inputs[(size_t)c * B_ * T_ + (size_t)b * T_ + t];
        }
        __syncthreads();
        #pragma unroll
        for (int r = 0; r < (B_ * TN) / NTHR; ++r) {   // 16 iters
            int e  = r * NTHR + tid;
            int b  = e >> 6;
            int nl = e & 63;
            int n  = nbase + nl;
            float ext = xls[0 * B_ + b] * enc[0 * (size_t)N_ + n]
                      + xls[1 * B_ + b] * enc[1 * (size_t)N_ + n]
                      + xls[2 * B_ + b] * enc[2 * (size_t)N_ + n]
                      + xls[3 * B_ + b] * enc[3 * (size_t)N_ + n];
            float u = sBuf[(size_t)b * N_ + n] + mask[n] * ext;
            aBuf[(size_t)b * N_ + n] = (__bf16)u;
        }
        // block 0 finalizes previous step's decode output & resets accumulator
        if (bid == 0 && t > 0) {
            int b = tid >> 2, c = tid & 3;   // tid == b*4+c
            float v = satclamp(dacc[tid] + dec_b[c]);
            out[(size_t)c * B_ * T_ + (size_t)b * T_ + (t - 1)] = v;
            dacc[tid] = 0.0f;
        }
        gbar(barCnt, barGen);

        // ================= Phase B: s_new = mask*(bias + clamp(u @ W^T)) ====
        sAcc[tid] = 0.0f;
        __syncthreads();

        // A (16x32 bf16): lane lr holds row m0+lr; hi selects K 0..15 vs 16..31 (contiguous in u row)
        // B (32x16 bf16): lane lr holds column n; hi selects K 0..15 vs 16..31 (contiguous in W row)
        const __bf16* aRow  = aBuf + (size_t)(m0 + lr) * N_;
        const __bf16* wRow0 = Wbf + (size_t)n0 * N_;
        const __bf16* wRow1 = Wbf + (size_t)n1 * N_;
        v8f acc0 = {};
        v8f acc1 = {};
        #pragma unroll 4
        for (int k0 = 0; k0 < N_; k0 += 32) {
            v16bf av = *(const v16bf*)(aRow + k0 + hi * 16);
            v16bf bv0 = *(const v16bf*)(wRow0 + k0 + hi * 16);
            v16bf bv1 = *(const v16bf*)(wRow1 + k0 + hi * 16);
            __builtin_prefetch(wRow0 + k0 + 256, 0, 1);
            __builtin_prefetch(wRow1 + k0 + 256, 0, 1);
            acc0 = __builtin_amdgcn_wmma_f32_16x16x32_bf16(
                       false, av, false, bv0, (short)0, acc0, false, false);
            acc1 = __builtin_amdgcn_wmma_f32_16x16x32_bf16(
                       false, av, false, bv1, (short)0, acc1, false, false);
        }

        // epilogue: clamp, bias, mask, store state, decode partials
        float mk0 = mask[n0],  mk1 = mask[n1];
        float bc0 = biasc[n0], bc1 = biasc[n1];
        float dw0[C_], dw1[C_];
        #pragma unroll
        for (int c = 0; c < C_; ++c) {
            dw0[c] = dec_w[(size_t)c * N_ + n0];
            dw1[c] = dec_w[(size_t)c * N_ + n1];
        }
        #pragma unroll
        for (int j = 0; j < 8; ++j) {
            int b = m0 + hi * 8 + j;             // D layout: lanes 0-15 -> M=j, 16-31 -> M=j+8
            float s0 = mk0 * (bc0 + satclamp(acc0[j]));
            float s1 = mk1 * (bc1 + satclamp(acc1[j]));
            sBuf[(size_t)b * N_ + n0] = s0;
            sBuf[(size_t)b * N_ + n1] = s1;
            #pragma unroll
            for (int c = 0; c < C_; ++c)
                atomicAdd(&sAcc[b * C_ + c], s0 * dw0[c] + s1 * dw1[c]);
        }
        __syncthreads();
        atomicAdd(&dacc[tid], sAcc[tid]);        // 256 entries, one per thread
        gbar(barCnt, barGen);
    }

    // finalize last step's decode
    if (bid == 0) {
        int b = tid >> 2, c = tid & 3;
        float v = satclamp(dacc[tid] + dec_b[c]);
        out[(size_t)c * B_ * T_ + (size_t)b * T_ + (T_ - 1)] = v;
        dacc[tid] = 0.0f;
    }
}

extern "C" void kernel_launch(void* const* d_in, const int* in_sizes, int n_in,
                              void* d_out, int out_size, void* d_ws, size_t ws_size,
                              hipStream_t stream) {
    const float* inputs = (const float*)d_in[0];   // [C,B,T]
    const float* state  = (const float*)d_in[1];   // [B,N]
    const float* mask   = (const float*)d_in[2];   // [N]
    const float* W      = (const float*)d_in[3];   // [N,N]
    const float* enc    = (const float*)d_in[4];   // [C,N]
    const float* biasc  = (const float*)d_in[5];   // [N]
    const float* dec_w  = (const float*)d_in[6];   // [C,N]
    const float* dec_b  = (const float*)d_in[7];   // [C]
    float* out = (float*)d_out;

    char* ws = (char*)d_ws;
    __bf16*   Wbf    = (__bf16*)(ws + OFF_WBF);
    __bf16*   aBuf   = (__bf16*)(ws + OFF_ABUF);
    float*    sBuf   = (float*)(ws + OFF_SBUF);
    float*    dacc   = (float*)(ws + OFF_DACC);
    unsigned* barCnt = (unsigned*)(ws + OFF_BAR);
    unsigned* barGen = barCnt + 1;

    rnn_init_kernel<<<1, 256, 0, stream>>>(dacc, barCnt, barGen);
    rnn_persistent_kernel<<<NBLK, NTHR, 0, stream>>>(
        inputs, state, mask, W, enc, biasc, dec_w, dec_b, out,
        Wbf, aBuf, sBuf, dacc, barCnt, barGen);
}
